// TrackMPNN_72885595013637
// MI455X (gfx1250) — matrix operand order repaired and checked by reading, not compiled, not run on hardware
//
#include <hip/hip_runtime.h>
#include <hip/hip_bf16.h>

#define BN_EPS 1e-5f

typedef __attribute__((ext_vector_type(16))) _Float16 v16h;
typedef __attribute__((ext_vector_type(8)))  float    v8f;

// ---------------------------------------------------------------------------
// WMMA helpers (CDNA5 wave32, V_WMMA_F32_16X16X32_F16)
// ---------------------------------------------------------------------------
__device__ __forceinline__ v8f wmma32(v16h a, v16h b, v8f c) {
  // 8 args: (neg_a, A, neg_b, B, c_mod, C, reuse_a, reuse_b)
  return __builtin_amdgcn_wmma_f32_16x16x32_f16(false, a, false, b, (short)0, c,
                                                false, false);
}

// B fragment: pre-swizzled; each lane reads 16 contiguous halfs (32B aligned).
// Works for both global and LDS pointers (addrspace inferred after inlining).
__device__ __forceinline__ v16h ldB(const _Float16* p, int tile, int lane) {
  return *(const v16h*)(p + ((size_t)tile << 9) + (lane << 4));
}

// A fragment (16x32 f16) from f32 row-major [nrows][64] matrix.
// lane<16: M=lane, halfs 0-7 -> K=kbase..+7, halfs 8-15 -> K=kbase+16..+23
// lane>=16: same rows with K offset +8.
__device__ __forceinline__ v16h ldA(const float* __restrict__ A, int row0, int nrows,
                                    int kbase, int lane) {
  int row = row0 + (lane & 15);
  if (row >= nrows) row = nrows - 1;       // clamp (no divergence before WMMA)
  int off = ((lane >> 4) & 1) * 8;
  const float* p = A + (size_t)row * 64 + kbase + off;
  float4 f0 = *(const float4*)(p);
  float4 f1 = *(const float4*)(p + 4);
  float4 f2 = *(const float4*)(p + 16);
  float4 f3 = *(const float4*)(p + 20);
  v16h a;
  a[0]  = (_Float16)f0.x; a[1]  = (_Float16)f0.y; a[2]  = (_Float16)f0.z; a[3]  = (_Float16)f0.w;
  a[4]  = (_Float16)f1.x; a[5]  = (_Float16)f1.y; a[6]  = (_Float16)f1.z; a[7]  = (_Float16)f1.w;
  a[8]  = (_Float16)f2.x; a[9]  = (_Float16)f2.y; a[10] = (_Float16)f2.z; a[11] = (_Float16)f2.w;
  a[12] = (_Float16)f3.x; a[13] = (_Float16)f3.y; a[14] = (_Float16)f3.z; a[15] = (_Float16)f3.w;
  return a;
}

// A fragment with fused BatchNorm affine + ReLU applied elementwise.
__device__ __forceinline__ v16h ldA_bnrelu(const float* __restrict__ T, int row0, int nrows,
                                           int kbase, int lane,
                                           const float* __restrict__ sc,
                                           const float* __restrict__ sh) {
  int row = row0 + (lane & 15);
  if (row >= nrows) row = nrows - 1;
  int off = ((lane >> 4) & 1) * 8;
  const float* p = T + (size_t)row * 64 + kbase + off;
  v16h a;
#pragma unroll
  for (int jj = 0; jj < 8; ++jj) {
    int k0 = kbase + off + jj;
    int k1 = kbase + 16 + off + jj;
    float v0 = fmaxf(p[jj]      * sc[k0] + sh[k0], 0.f);
    float v1 = fmaxf(p[16 + jj] * sc[k1] + sh[k1], 0.f);
    a[jj]     = (_Float16)v0;
    a[jj + 8] = (_Float16)v1;
  }
  return a;
}

// ---------------------------------------------------------------------------
// Fully-fused GRU cell for one 16-row x 16-feature block.
// r/z accumulators are seeded with bi+bh and accumulate BOTH m@Wi^T and h@Wh^T
// (reference computes sigmoid(ir+hr) etc., so fusion is exact).
// B fragments come from LDS (async-staged).
// ---------------------------------------------------------------------------
__device__ __forceinline__ v8f gru_block(v16h am0, v16h am1, v16h ah0, v16h ah1,
                                         const _Float16* pWi, const _Float16* pWh,
                                         const float* __restrict__ bi,
                                         const float* __restrict__ bh,
                                         v8f hfrag, int f, int j, int lane) {
  const int NT = 12;  // 192 gate columns / 16
  float b;

  b = bi[f] + bh[f];                       // r gate, cols [0,64)
  v8f ar = {b, b, b, b, b, b, b, b};
  ar = wmma32(am0, ldB(pWi, 0 * NT + j, lane), ar);
  ar = wmma32(am1, ldB(pWi, 1 * NT + j, lane), ar);
  ar = wmma32(ah0, ldB(pWh, 0 * NT + j, lane), ar);
  ar = wmma32(ah1, ldB(pWh, 1 * NT + j, lane), ar);

  b = bi[64 + f] + bh[64 + f];             // z gate, cols [64,128)
  v8f az = {b, b, b, b, b, b, b, b};
  az = wmma32(am0, ldB(pWi, 0 * NT + 4 + j, lane), az);
  az = wmma32(am1, ldB(pWi, 1 * NT + 4 + j, lane), az);
  az = wmma32(ah0, ldB(pWh, 0 * NT + 4 + j, lane), az);
  az = wmma32(ah1, ldB(pWh, 1 * NT + 4 + j, lane), az);

  b = bi[128 + f];                         // inn, cols [128,192)
  v8f ai = {b, b, b, b, b, b, b, b};
  ai = wmma32(am0, ldB(pWi, 0 * NT + 8 + j, lane), ai);
  ai = wmma32(am1, ldB(pWi, 1 * NT + 8 + j, lane), ai);

  b = bh[128 + f];                         // hn
  v8f an = {b, b, b, b, b, b, b, b};
  an = wmma32(ah0, ldB(pWh, 0 * NT + 8 + j, lane), an);
  an = wmma32(ah1, ldB(pWh, 1 * NT + 8 + j, lane), an);

  v8f out;
#pragma unroll
  for (int s = 0; s < 8; ++s) {
    float r  = 1.f / (1.f + expf(-ar[s]));
    float z  = 1.f / (1.f + expf(-az[s]));
    float ng = tanhf(ai[s] + r * an[s]);
    out[s]   = (1.f - z) * ng + z * hfrag[s];
  }
  return out;
}

// ---------------------------------------------------------------------------
// Kernels
// ---------------------------------------------------------------------------
__global__ void zero_f32_kernel(float* __restrict__ p, long long n) {
  long long i = (long long)blockIdx.x * blockDim.x + threadIdx.x;
  if (i < n) p[i] = 0.f;
}

__global__ void copy_f32_kernel(float* __restrict__ dst, const float* __restrict__ src,
                                long long n) {
  long long i = (long long)blockIdx.x * blockDim.x + threadIdx.x;
  if (i < n) dst[i] = src[i];
}

// Diagonal extraction: d[r] += vals where rows==cols.
__global__ void diag_kernel(const int* __restrict__ rows, const int* __restrict__ cols,
                            const float* __restrict__ vals, float* __restrict__ d, int nnz) {
  int i = blockIdx.x * blockDim.x + threadIdx.x;
  if (i < nnz) {
    int r = rows[i];
    if (r == cols[i]) atomicAdd(&d[r], vals[i]);
  }
}

// Pack W[ngate][K] (f32) into CDNA5 B-fragment layout (f16), 32x16 tiles.
// tile = kt*(ngate/16)+nt; lane<16: N=nt*16+lane, j<8: K=kt*32+j, j>=8: K=kt*32+16+(j-8); lane>=16: K +8.
__global__ void pack_b_kernel(const float* __restrict__ W, _Float16* __restrict__ packed,
                              int ngate, int K) {
  int idx = blockIdx.x * blockDim.x + threadIdx.x;
  int total = ngate * K;
  if (idx >= total) return;
  int tile = idx >> 9;
  int e    = idx & 511;
  int lane = e >> 4;
  int jj   = e & 15;
  int ntiles = ngate >> 4;
  int kt = tile / ntiles;
  int nt = tile - kt * ntiles;
  int n  = nt * 16 + (lane & 15);
  int k  = kt * 32 + ((jj >> 3) << 4) + ((lane >> 4) << 3) + (jj & 7);
  packed[idx] = (_Float16)W[(size_t)n * K + k];
}

// SpMM scatter: 2 nonzeros per wave (16 lanes x float4 each); skip diagonal.
// h (102 MB) is L2-resident on MI455X (192 MB L2) so gathers mostly hit L2.
__global__ void __launch_bounds__(256) spmm_kernel(const int* __restrict__ rows,
                                                   const int* __restrict__ cols,
                                                   const float* __restrict__ vals,
                                                   const float* __restrict__ h,
                                                   float* __restrict__ m, int nnz) {
  int wid  = blockIdx.x * 8 + (threadIdx.x >> 5);
  int lane = threadIdx.x & 31;
  int e = wid * 2 + ((lane >> 4) & 1);
  if (e >= nnz) return;
  int r = rows[e], c = cols[e];
  if (r != c) {                            // off-diagonal only
    float v = vals[e];
    int fo = (lane & 15) * 4;
    float4 hv = *(const float4*)(h + (size_t)c * 64 + fo);
    float* mp = m + (size_t)r * 64 + fo;
    atomicAdd(mp + 0, v * hv.x);
    atomicAdd(mp + 1, v * hv.y);
    atomicAdd(mp + 2, v * hv.z);
    atomicAdd(mp + 3, v * hv.w);
  }
}

// t = x @ W1^T + b1 ; also accumulate per-column sum / sumsq for BatchNorm.
__global__ void __launch_bounds__(256) gemm_in1_kernel(
    const float* __restrict__ x, const _Float16* __restrict__ pW1,
    const float* __restrict__ b1, float* __restrict__ t,
    float* __restrict__ bnsum, float* __restrict__ bnsumsq, int Nx) {
  int lane = threadIdx.x & 31;
  int tile = blockIdx.x * 8 + (threadIdx.x >> 5);
  int row0 = tile * 16;
  if (row0 >= Nx) return;
  int hi = (lane >> 4) & 1;
  int cb = lane & 15;
  bool full = (row0 + 16 <= Nx);           // wave-uniform: no EXEC churn
  v16h a0 = ldA(x, row0, Nx, 0,  lane);
  v16h a1 = ldA(x, row0, Nx, 32, lane);
#pragma unroll
  for (int nt = 0; nt < 4; ++nt) {
    int col = nt * 16 + cb;
    float b = b1[col];
    v8f acc = {b, b, b, b, b, b, b, b};
    acc = wmma32(a0, ldB(pW1, 0 * 4 + nt, lane), acc);
    acc = wmma32(a1, ldB(pW1, 1 * 4 + nt, lane), acc);
    float s = 0.f, sq = 0.f;
    if (full) {
#pragma unroll
      for (int k = 0; k < 8; ++k) {
        t[(size_t)(row0 + k + hi * 8) * 64 + col] = acc[k];
        s += acc[k]; sq += acc[k] * acc[k];
      }
    } else {
#pragma unroll
      for (int k = 0; k < 8; ++k) {
        int row = row0 + k + hi * 8;
        if (row < Nx) {
          t[(size_t)row * 64 + col] = acc[k];
          s += acc[k]; sq += acc[k] * acc[k];
        }
      }
    }
    // lanes L and L^16 share the same output column -> fold, then 1 atomic/col
    s  += __shfl_xor(s, 16, 32);
    sq += __shfl_xor(sq, 16, 32);
    if (lane < 16) {
      atomicAdd(&bnsum[col], s);
      atomicAdd(&bnsumsq[col], sq);
    }
  }
}

__global__ void bn_finalize_kernel(const float* __restrict__ bnsum,
                                   const float* __restrict__ bnsumsq,
                                   const float* __restrict__ gamma,
                                   const float* __restrict__ beta,
                                   float* __restrict__ scale, float* __restrict__ shift,
                                   int Nx) {
  int i = threadIdx.x;
  if (i < 64) {
    float inv_n = 1.f / (float)Nx;
    float mu  = bnsum[i] * inv_n;
    float var = bnsumsq[i] * inv_n - mu * mu;   // biased var (torch BN forward)
    float istd = rsqrtf(var + BN_EPS);
    float sc = istd * gamma[i];
    scale[i] = sc;
    shift[i] = beta[i] - mu * sc;
  }
}

// h[Nold+row] = dn[Nold+row] * (relu(BN(t)) @ W2^T + b2)   (BN+ReLU fused into A load)
__global__ void __launch_bounds__(256) gemm_in2_kernel(
    const float* __restrict__ t, const _Float16* __restrict__ pW2,
    const float* __restrict__ b2, const float* __restrict__ scale,
    const float* __restrict__ shift, const float* __restrict__ dn,
    float* __restrict__ h, int Nx, int Nold) {
  int lane = threadIdx.x & 31;
  int tile = blockIdx.x * 8 + (threadIdx.x >> 5);
  int row0 = tile * 16;
  if (row0 >= Nx) return;
  int hi = (lane >> 4) & 1;
  int cb = lane & 15;
  bool full = (row0 + 16 <= Nx);
  v16h a0 = ldA_bnrelu(t, row0, Nx, 0,  lane, scale, shift);
  v16h a1 = ldA_bnrelu(t, row0, Nx, 32, lane, scale, shift);
#pragma unroll
  for (int nt = 0; nt < 4; ++nt) {
    int col = nt * 16 + cb;
    float b = b2[col];
    v8f acc = {b, b, b, b, b, b, b, b};
    acc = wmma32(a0, ldB(pW2, 0 * 4 + nt, lane), acc);
    acc = wmma32(a1, ldB(pW2, 1 * 4 + nt, lane), acc);
    if (full) {
#pragma unroll
      for (int s = 0; s < 8; ++s) {
        int g = Nold + row0 + s + hi * 8;
        h[(size_t)g * 64 + col] = dn[g] * acc[s];
      }
    } else {
#pragma unroll
      for (int s = 0; s < 8; ++s) {
        int row = row0 + s + hi * 8;
        if (row < Nx) {
          int g = Nold + row;
          h[(size_t)g * 64 + col] = dn[g] * acc[s];
        }
      }
    }
  }
}

// ---------------------------------------------------------------------------
// Fused dual-GRU + combine + output heads. One wave per 16-row tile.
// All four packed GRU weight matrices (96 KB) are async-staged into LDS once
// per workgroup via GLOBAL_LOAD_ASYNC_TO_LDS_B128 (ASYNCcnt-tracked), then B
// fragments are read as aligned ds_load_b128.
// ---------------------------------------------------------------------------
#define WSEG_HALFS 12288   // 24 tiles * 512 halfs
#define WSEG_BYTES 24576

__global__ void __launch_bounds__(256) gru_fused_kernel(
    const float* __restrict__ m_n, const float* __restrict__ m_e,
    const float* __restrict__ h,
    const _Float16* __restrict__ pWi_n, const _Float16* __restrict__ pWh_n,
    const _Float16* __restrict__ pWi_e, const _Float16* __restrict__ pWh_e,
    const float* __restrict__ bi_n, const float* __restrict__ bh_n,
    const float* __restrict__ bi_e, const float* __restrict__ bh_e,
    const float* __restrict__ dn, const float* __restrict__ de,
    const float* __restrict__ w_on, const float* __restrict__ b_on,
    const float* __restrict__ w_oe, const float* __restrict__ b_oe,
    float* __restrict__ out_sig, float* __restrict__ out_y,
    float* __restrict__ h_out, int N) {
  extern __shared__ _Float16 smem[];       // 4 * WSEG_HALFS halfs = 98304 B

  // --- cooperative async staging (ALL threads participate; before any exit) ---
  {
    const _Float16* srcs[4] = {pWi_n, pWh_n, pWi_e, pWh_e};
    unsigned lbase = (unsigned)(uintptr_t)smem;   // LDS byte offset (addr[31:0])
#pragma unroll
    for (int seg = 0; seg < 4; ++seg) {
      unsigned long long gbase = (unsigned long long)srcs[seg];
      for (int i = threadIdx.x; i < WSEG_BYTES / 16; i += 256) {  // uniform trip
        unsigned loff = lbase + (unsigned)seg * WSEG_BYTES + (unsigned)i * 16u;
        unsigned goff = (unsigned)i * 16u;
        asm volatile("global_load_async_to_lds_b128 %0, %1, %2"
                     :: "v"(loff), "v"(goff), "s"(gbase)
                     : "memory");
      }
    }
    asm volatile("s_wait_asynccnt 0" ::: "memory");
    __syncthreads();
  }
  const _Float16* sWi_n = smem;
  const _Float16* sWh_n = smem + 1 * WSEG_HALFS;
  const _Float16* sWi_e = smem + 2 * WSEG_HALFS;
  const _Float16* sWh_e = smem + 3 * WSEG_HALFS;

  int lane = threadIdx.x & 31;
  int tile = blockIdx.x * 8 + (threadIdx.x >> 5);
  int row0 = tile * 16;
  if (row0 >= N) return;
  int hi = (lane >> 4) & 1;
  int cb = lane & 15;
  bool full = (row0 + 16 <= N);            // wave-uniform

  // A fragments for this row tile: messages + hidden (K=64 -> 2 frags each)
  v16h an0 = ldA(m_n, row0, N, 0, lane), an1 = ldA(m_n, row0, N, 32, lane);
  v16h ae0 = ldA(m_e, row0, N, 0, lane), ae1 = ldA(m_e, row0, N, 32, lane);
  v16h ah0 = ldA(h,   row0, N, 0, lane), ah1 = ldA(h,   row0, N, 32, lane);

  float dnv[8], dev[8];
#pragma unroll
  for (int s = 0; s < 8; ++s) {
    int row = row0 + s + hi * 8;
    if (row >= N) row = N - 1;
    dnv[s] = dn[row];
    dev[s] = de[row];
  }

  float pon[8], poe[8];
#pragma unroll
  for (int s = 0; s < 8; ++s) { pon[s] = 0.f; poe[s] = 0.f; }

#pragma unroll
  for (int j = 0; j < 4; ++j) {            // 4 blocks of 16 hidden features
    int f = j * 16 + cb;
    v8f hfrag;                             // h tile in C-fragment layout
    if (full) {
#pragma unroll
      for (int s = 0; s < 8; ++s)
        hfrag[s] = h[(size_t)(row0 + s + hi * 8) * 64 + f];
    } else {
#pragma unroll
      for (int s = 0; s < 8; ++s) {
        int row = row0 + s + hi * 8;
        if (row >= N) row = N - 1;
        hfrag[s] = h[(size_t)row * 64 + f];
      }
    }
    v8f hnode = gru_block(an0, an1, ah0, ah1, sWi_n, sWh_n, bi_n, bh_n, hfrag, f, j, lane);
    v8f hedge = gru_block(ae0, ae1, ah0, ah1, sWi_e, sWh_e, bi_e, bh_e, hfrag, f, j, lane);
    float won = w_on[f], woe = w_oe[f];
    if (full) {
#pragma unroll
      for (int s = 0; s < 8; ++s) {
        float ho = dnv[s] * hnode[s] + dev[s] * hedge[s];
        h_out[(size_t)(row0 + s + hi * 8) * 64 + f] = ho;
        pon[s] += ho * won;
        poe[s] += ho * woe;
      }
    } else {
#pragma unroll
      for (int s = 0; s < 8; ++s) {
        float ho = dnv[s] * hnode[s] + dev[s] * hedge[s];
        int row = row0 + s + hi * 8;
        if (row < N) h_out[(size_t)row * 64 + f] = ho;
        pon[s] += ho * won;
        poe[s] += ho * woe;
      }
    }
  }

  // Output heads: reduce dot-products over the 16 lanes sharing each row.
  float bon = b_on[0], boe = b_oe[0];
#pragma unroll
  for (int s = 0; s < 8; ++s) {
    float a = pon[s], b = poe[s];
#pragma unroll
    for (int off = 8; off >= 1; off >>= 1) {
      a += __shfl_xor(a, off, 16);
      b += __shfl_xor(b, off, 16);
    }
    if (cb == 0) {
      int row = row0 + s + hi * 8;
      if (row < N) {
        float yv = dnv[s] * (a + bon) + dev[s] * (b + boe);
        out_y[row]   = yv;
        out_sig[row] = 1.f / (1.f + expf(-yv));
      }
    }
  }
}

// ---------------------------------------------------------------------------
// Host launcher
// ---------------------------------------------------------------------------
extern "C" void kernel_launch(void* const* d_in, const int* in_sizes, int n_in,
                              void* d_out, int out_size, void* d_ws, size_t ws_size,
                              hipStream_t stream) {
  const float* x     = (const float*)d_in[0];
  const float* h_in  = (const float*)d_in[1];
  const int*   nrows = (const int*)d_in[2];
  const int*   ncols = (const int*)d_in[3];
  const float* nvals = (const float*)d_in[4];
  const int*   erows = (const int*)d_in[5];
  const int*   ecols = (const int*)d_in[6];
  const float* evals = (const float*)d_in[7];
  const float* W1    = (const float*)d_in[8];
  const float* b1    = (const float*)d_in[9];
  const float* gamma = (const float*)d_in[10];
  const float* beta  = (const float*)d_in[11];
  const float* W2    = (const float*)d_in[12];
  const float* b2    = (const float*)d_in[13];
  const float* Wi_n  = (const float*)d_in[14];
  const float* bi_n  = (const float*)d_in[15];
  const float* Wh_n  = (const float*)d_in[16];
  const float* bh_n  = (const float*)d_in[17];
  const float* Wi_e  = (const float*)d_in[18];
  const float* bi_e  = (const float*)d_in[19];
  const float* Wh_e  = (const float*)d_in[20];
  const float* bh_e  = (const float*)d_in[21];
  const float* w_on  = (const float*)d_in[22];
  const float* b_on  = (const float*)d_in[23];
  const float* w_oe  = (const float*)d_in[24];
  const float* b_oe  = (const float*)d_in[25];

  const int Nx    = in_sizes[0] / 64;
  const int Nold  = in_sizes[1] / 64;
  const int N     = Nx + Nold;
  const int nnz_n = in_sizes[2];
  const int nnz_e = in_sizes[5];

  // Workspace carve-out (256B aligned)
  char* ws = (char*)d_ws;
  size_t off = 0;
  auto alloc = [&](size_t bytes) {
    void* p = ws + off;
    off = (off + bytes + 255) & ~(size_t)255;
    return p;
  };
  float* dn      = (float*)alloc((size_t)N * 4);
  float* de      = (float*)alloc((size_t)N * 4);
  float* hbuf    = (float*)alloc((size_t)N * 64 * 4);
  float* m_n     = (float*)alloc((size_t)N * 64 * 4);
  float* m_e     = (float*)alloc((size_t)N * 64 * 4);
  float* tbuf    = (float*)alloc((size_t)Nx * 64 * 4);
  float* bnsum   = (float*)alloc(64 * 4);
  float* bnsumsq = (float*)alloc(64 * 4);
  float* scale   = (float*)alloc(64 * 4);
  float* shift   = (float*)alloc(64 * 4);
  _Float16* pW1   = (_Float16*)alloc(8  * 512 * 2);
  _Float16* pW2   = (_Float16*)alloc(8  * 512 * 2);
  _Float16* pWi_n = (_Float16*)alloc(24 * 512 * 2);
  _Float16* pWh_n = (_Float16*)alloc(24 * 512 * 2);
  _Float16* pWi_e = (_Float16*)alloc(24 * 512 * 2);
  _Float16* pWh_e = (_Float16*)alloc(24 * 512 * 2);

  float* out_sig = (float*)d_out;          // sigmoid(y): N
  float* out_y   = out_sig + N;            // y: N
  float* h_out   = out_y + N;              // h_out: N x 64

  auto cdiv = [](long long a, long long b) { return (int)((a + b - 1) / b); };

  // 1) zero accumulators (ws is poisoned by harness; must re-zero every call)
  zero_f32_kernel<<<cdiv(N, 256), 256, 0, stream>>>(dn, N);
  zero_f32_kernel<<<cdiv(N, 256), 256, 0, stream>>>(de, N);
  zero_f32_kernel<<<cdiv((long long)N * 64, 256), 256, 0, stream>>>(m_n, (long long)N * 64);
  zero_f32_kernel<<<cdiv((long long)N * 64, 256), 256, 0, stream>>>(m_e, (long long)N * 64);
  zero_f32_kernel<<<1, 256, 0, stream>>>(bnsum, 64);
  zero_f32_kernel<<<1, 256, 0, stream>>>(bnsumsq, 64);

  // 2) diagonal masks dn/de
  diag_kernel<<<cdiv(nnz_n, 256), 256, 0, stream>>>(nrows, ncols, nvals, dn, nnz_n);
  diag_kernel<<<cdiv(nnz_e, 256), 256, 0, stream>>>(erows, ecols, evals, de, nnz_e);

  // 3) pack all weights into WMMA B-fragment layout (f16)
  pack_b_kernel<<<cdiv(64  * 64, 256), 256, 0, stream>>>(W1,   pW1,   64,  64);
  pack_b_kernel<<<cdiv(64  * 64, 256), 256, 0, stream>>>(W2,   pW2,   64,  64);
  pack_b_kernel<<<cdiv(192 * 64, 256), 256, 0, stream>>>(Wi_n, pWi_n, 192, 64);
  pack_b_kernel<<<cdiv(192 * 64, 256), 256, 0, stream>>>(Wh_n, pWh_n, 192, 64);
  pack_b_kernel<<<cdiv(192 * 64, 256), 256, 0, stream>>>(Wi_e, pWi_e, 192, 64);
  pack_b_kernel<<<cdiv(192 * 64, 256), 256, 0, stream>>>(Wh_e, pWh_e, 192, 64);

  // 4) h[0:Nold] = h_in
  copy_f32_kernel<<<cdiv((long long)Nold * 64, 256), 256, 0, stream>>>(
      hbuf, h_in, (long long)Nold * 64);

  // 5) input transform: GEMM1 (+BN stats) -> BN finalize -> GEMM2 (+mask into h tail)
  int tilesX = (Nx + 15) / 16;
  gemm_in1_kernel<<<(tilesX + 7) / 8, 256, 0, stream>>>(x, pW1, b1, tbuf, bnsum, bnsumsq, Nx);
  bn_finalize_kernel<<<1, 64, 0, stream>>>(bnsum, bnsumsq, gamma, beta, scale, shift, Nx);
  gemm_in2_kernel<<<(tilesX + 7) / 8, 256, 0, stream>>>(tbuf, pW2, b2, scale, shift, dn,
                                                        hbuf, Nx, Nold);

  // 6) sparse message passing (off-diagonal scatter; h is L2-resident)
  spmm_kernel<<<cdiv(cdiv(nnz_n, 2) * 32LL, 256), 256, 0, stream>>>(nrows, ncols, nvals,
                                                                    hbuf, m_n, nnz_n);
  spmm_kernel<<<cdiv(cdiv(nnz_e, 2) * 32LL, 256), 256, 0, stream>>>(erows, ecols, evals,
                                                                    hbuf, m_e, nnz_e);

  // 7) fused dual-GRU + combine + output heads (96 WMMAs per 16-row tile);
  //    98304 B dynamic LDS for the async-staged weight fragments.
  int tilesN = (N + 15) / 16;
  gru_fused_kernel<<<(tilesN + 7) / 8, 256, 4 * WSEG_BYTES, stream>>>(
      m_n, m_e, hbuf, pWi_n, pWh_n, pWi_e, pWh_e,
      bi_n, bh_n, bi_e, bh_e, dn, de,
      w_on, b_on, w_oe, b_oe,
      out_sig, out_y, h_out, N);

  (void)n_in; (void)out_size; (void)ws_size;
}
